// FeedbackModelPseudo_74311524155789
// MI455X (gfx1250) — compile-verified, matrix operand order checked by reading, not run on hardware
//
#include <hip/hip_runtime.h>
#include <hip/hip_bf16.h>

// ---------------------------------------------------------------------------
// Problem constants (deberta-v3-large span head)
// ---------------------------------------------------------------------------
#define BB 16
#define TT 1024
#define HID 1024
#define SS 32
#define NHEAD 16
#define DHEAD 64
#define HHALF 512        // LSTM hidden per direction
#define G4 2048          // 4*HHALF
#define NL 3
#define LN_EPS 1e-7f

typedef __attribute__((ext_vector_type(16))) __bf16 bf16x16;
typedef __attribute__((ext_vector_type(8)))  float  f32x8;

#if defined(__has_builtin)
#if __has_builtin(__builtin_amdgcn_tensor_load_to_lds) && __has_builtin(__builtin_amdgcn_s_wait_tensorcnt)
#define HAVE_TDM 1
#endif
#endif

#ifdef HAVE_TDM
typedef unsigned int u32x4 __attribute__((ext_vector_type(4)));
typedef int          i32x4 __attribute__((ext_vector_type(4)));
typedef int          i32x8 __attribute__((ext_vector_type(8)));
#endif

union FragAB { bf16x16 v; int4 q[2]; };

// Load a 16x32 (A) or 32x16-as-[N,K] (B) bf16 WMMA fragment: two contiguous
// 16B chunks per lane at (row, kb) and (row, kb+16), kb = (lane>>4)*8.
__device__ __forceinline__ bf16x16 load_frag(const __bf16* __restrict__ p,
                                             int ld, int row, int k0, int kb) {
    FragAB f;
    const __bf16* base = p + (size_t)row * ld + k0 + kb;
    f.q[0] = *(const int4*)(base);
    f.q[1] = *(const int4*)(base + 16);
    return f.v;
}

__device__ __forceinline__ float sigmoidf_(float x) {
    return 1.0f / (1.0f + __expf(-x));
}

// ---------------------------------------------------------------------------
// f32 -> bf16 conversion (grid-stride)
// ---------------------------------------------------------------------------
__global__ void k_f32_to_bf16(const float* __restrict__ s, __bf16* __restrict__ d, long n) {
    long i = (long)blockIdx.x * blockDim.x + threadIdx.x;
    long stride = (long)gridDim.x * blockDim.x;
    for (; i < n; i += stride) d[i] = (__bf16)s[i];
}

// ---------------------------------------------------------------------------
// Dense GEMM: C = A[M,K](bf16) * B[N,K](bf16, W row-major) + bias[N]
// block = 128 threads = 4 waves; each wave owns a 32(M) x 64(N) tile
// (2 A-frags, 4 B-frags, 8 accumulators -> 8 WMMAs per 12 b128 loads).
// K-loop is register software-pipelined: fragments for k+32 are in flight
// while the 8 WMMAs for k execute (partial s_wait_loadcnt).
// permute_rows: store C row m as (m%T)*B + m/T  -> writes xg in [T,B,4096]
// layout so the LSTM recurrence reads one contiguous block per step.
// ---------------------------------------------------------------------------
__global__ __launch_bounds__(128)
void k_wmma_gemm(const __bf16* __restrict__ A, const __bf16* __restrict__ B,
                 const float* __restrict__ bias, float* __restrict__ C,
                 int M, int N, int K, int permute_rows) {
    const int lane = threadIdx.x & 31;
    const int wave = threadIdx.x >> 5;
    const int r    = lane & 15;
    const int half = lane >> 4;
    const int kb   = half * 8;
    const int m0   = blockIdx.y * 32;
    const int n0   = blockIdx.x * 256 + wave * 64;

    f32x8 acc[2][4];
#pragma unroll
    for (int mi = 0; mi < 2; ++mi)
#pragma unroll
        for (int j = 0; j < 4; ++j)
#pragma unroll
            for (int v = 0; v < 8; ++v) acc[mi][j][v] = 0.0f;

    // pipeline prologue: fragments for k0 = 0
    bf16x16 ca0 = load_frag(A, K, m0 + r,      0, kb);
    bf16x16 ca1 = load_frag(A, K, m0 + 16 + r, 0, kb);
    bf16x16 cb0 = load_frag(B, K, n0 +  0 + r, 0, kb);
    bf16x16 cb1 = load_frag(B, K, n0 + 16 + r, 0, kb);
    bf16x16 cb2 = load_frag(B, K, n0 + 32 + r, 0, kb);
    bf16x16 cb3 = load_frag(B, K, n0 + 48 + r, 0, kb);

    for (int k0 = 0; k0 < K - 32; k0 += 32) {
        const int kn = k0 + 32;
        // issue next-iteration loads before consuming current fragments
        bf16x16 na0 = load_frag(A, K, m0 + r,      kn, kb);
        bf16x16 na1 = load_frag(A, K, m0 + 16 + r, kn, kb);
        bf16x16 nb0 = load_frag(B, K, n0 +  0 + r, kn, kb);
        bf16x16 nb1 = load_frag(B, K, n0 + 16 + r, kn, kb);
        bf16x16 nb2 = load_frag(B, K, n0 + 32 + r, kn, kb);
        bf16x16 nb3 = load_frag(B, K, n0 + 48 + r, kn, kb);

        acc[0][0] = __builtin_amdgcn_wmma_f32_16x16x32_bf16(false, ca0, false, cb0, (short)0, acc[0][0], false, false);
        acc[0][1] = __builtin_amdgcn_wmma_f32_16x16x32_bf16(false, ca0, false, cb1, (short)0, acc[0][1], false, false);
        acc[0][2] = __builtin_amdgcn_wmma_f32_16x16x32_bf16(false, ca0, false, cb2, (short)0, acc[0][2], false, false);
        acc[0][3] = __builtin_amdgcn_wmma_f32_16x16x32_bf16(false, ca0, false, cb3, (short)0, acc[0][3], false, false);
        acc[1][0] = __builtin_amdgcn_wmma_f32_16x16x32_bf16(false, ca1, false, cb0, (short)0, acc[1][0], false, false);
        acc[1][1] = __builtin_amdgcn_wmma_f32_16x16x32_bf16(false, ca1, false, cb1, (short)0, acc[1][1], false, false);
        acc[1][2] = __builtin_amdgcn_wmma_f32_16x16x32_bf16(false, ca1, false, cb2, (short)0, acc[1][2], false, false);
        acc[1][3] = __builtin_amdgcn_wmma_f32_16x16x32_bf16(false, ca1, false, cb3, (short)0, acc[1][3], false, false);

        ca0 = na0; ca1 = na1; cb0 = nb0; cb1 = nb1; cb2 = nb2; cb3 = nb3;
    }
    // pipeline epilogue: last k-step
    acc[0][0] = __builtin_amdgcn_wmma_f32_16x16x32_bf16(false, ca0, false, cb0, (short)0, acc[0][0], false, false);
    acc[0][1] = __builtin_amdgcn_wmma_f32_16x16x32_bf16(false, ca0, false, cb1, (short)0, acc[0][1], false, false);
    acc[0][2] = __builtin_amdgcn_wmma_f32_16x16x32_bf16(false, ca0, false, cb2, (short)0, acc[0][2], false, false);
    acc[0][3] = __builtin_amdgcn_wmma_f32_16x16x32_bf16(false, ca0, false, cb3, (short)0, acc[0][3], false, false);
    acc[1][0] = __builtin_amdgcn_wmma_f32_16x16x32_bf16(false, ca1, false, cb0, (short)0, acc[1][0], false, false);
    acc[1][1] = __builtin_amdgcn_wmma_f32_16x16x32_bf16(false, ca1, false, cb1, (short)0, acc[1][1], false, false);
    acc[1][2] = __builtin_amdgcn_wmma_f32_16x16x32_bf16(false, ca1, false, cb2, (short)0, acc[1][2], false, false);
    acc[1][3] = __builtin_amdgcn_wmma_f32_16x16x32_bf16(false, ca1, false, cb3, (short)0, acc[1][3], false, false);

#pragma unroll
    for (int mi = 0; mi < 2; ++mi)
#pragma unroll
        for (int j = 0; j < 4; ++j) {
            const int n = n0 + j * 16 + r;
            const float bs = bias ? bias[n] : 0.0f;
#pragma unroll
            for (int v = 0; v < 8; ++v) {
                const int m = m0 + mi * 16 + half * 8 + v;
                const int mrow = permute_rows ? ((m & (TT - 1)) * BB + (m >> 10)) : m;
                C[(size_t)mrow * N + n] = acc[mi][j][v] + bs;
            }
        }
}

// ---------------------------------------------------------------------------
// Persistent bi-LSTM recurrence.
// grid = (32 shards, 2 directions), block = 128 (4 waves).
// Shard s owns hidden units [s*16, s*16+16); its w_hh slice (4 gates x 16
// units x 512 K, bf16 = 64 KB) is LDS-resident for all 1024 steps.
// The shard is staged by the Tensor Data Mover (4 contiguous 16 KB gate
// blocks, 1-D descriptors, TENSORcnt-tracked) when the builtin is available;
// otherwise by plain vector loads. h (16x512 bf16 per dir) is
// double-buffered in global (L2-resident). Cell state c lives in registers
// (maps 1:1 onto the WMMA D layout). xg is read in [T, B, 4096] layout and
// its loads are issued BEFORE the WMMA k-loop so they hide under matrix ops.
// ---------------------------------------------------------------------------
__global__ __launch_bounds__(128)
void k_lstm_recurrence(const float* __restrict__ xg,     // [T, B, 4096]
                       const __bf16* __restrict__ whh,   // [2, 2048, 512] bf16
                       __bf16* __restrict__ hbf,         // [2 dir][2 buf][16][512]
                       float* __restrict__ enc,          // [B, T, 1024]
                       int* __restrict__ bar) {          // [2 dirs x 64-int stride]
    __shared__ int4 wl4[64 * 64];                        // 64 rows x 512 bf16 = 64 KB
    __bf16* wl = (__bf16*)wl4;

    const int d    = blockIdx.y;       // 0 = fwd, 1 = bwd
    const int s    = blockIdx.x;       // 0..31
    const int j0   = s * 16;
    const int tid  = threadIdx.x;
    const int lane = tid & 31;
    const int r    = lane & 15;
    const int half = lane >> 4;
    const int kb   = half * 8;

#ifdef HAVE_TDM
    // Stage w_hh shard into LDS with the Tensor Data Mover. Each gate block
    // (rows g*512+j0 .. +15, full 512-col rows) is one contiguous 16 KB chunk
    // = a 1-D tile of 8192 bf16 elements. Wave 0 issues 4 descriptors.
    if (tid < 32) {
#pragma unroll
        for (int g = 0; g < 4; ++g) {
            const unsigned lds_off =
                (unsigned)(uintptr_t)wl4 + (unsigned)(g * 16 * 512 * 2);
            const unsigned long long ga = (unsigned long long)(uintptr_t)
                (whh + ((size_t)d * G4 + g * 512 + j0) * HHALF);
            // D# group 0: count=1 | lds_addr | global_addr | type=2 ("image")
            u32x4 g0 = { 1u,
                         lds_off,
                         (unsigned)(ga & 0xffffffffu),
                         (unsigned)((ga >> 32) & 0x01ffffffu) | (2u << 30) };
            // D# group 1: mask=0 (not in cluster), data_size=1 (2B),
            // tensor_dim0 = tile_dim0 = 8192, tensor_dim1 = tile_dim1 = 1,
            // tensor_dim0_stride = 8192.
            i32x8 g1 = { (int)(1u << 16),
                         (int)(8192u << 16),
                         (int)(1u << 16),
                         (int)(8192u << 16),
                         1,
                         8192,
                         0, 0 };
            i32x4 g2 = { 0, 0, 0, 0 };
            i32x4 g3 = { 0, 0, 0, 0 };
#if __clang_major__ >= 23
            i32x8 g4 = { 0, 0, 0, 0, 0, 0, 0, 0 };
            __builtin_amdgcn_tensor_load_to_lds(g0, g1, g2, g3, g4, 0);
#else
            __builtin_amdgcn_tensor_load_to_lds(g0, g1, g2, g3, 0);
#endif
        }
        __builtin_amdgcn_s_wait_tensorcnt(0);
    }
#else
    // Fallback: plain vector-load staging. LDS row rr = gate*16 + col;
    // global row = gate*512 + j0 + col. Rows are 64 int4 wide.
    for (int i = tid; i < 64 * 64; i += 128) {
        const int rr = i >> 6, c = i & 63;
        const int grow = (rr >> 4) * 512 + j0 + (rr & 15);
        wl4[i] = ((const int4*)(whh + ((size_t)d * G4 + grow) * HHALF))[c];
    }
#endif

    // Zero both h buffers for our unit slice (h0 = 0).
    for (int i = tid; i < 2 * 16 * 16; i += 128) {
        const int buf = i >> 8, m = (i >> 4) & 15, j = j0 + (i & 15);
        hbf[((size_t)(d * 2 + buf) * 16 + m) * HHALF + j] = (__bf16)0.0f;
    }

    float c[8];
#pragma unroll
    for (int v = 0; v < 8; ++v) c[v] = 0.0f;

    volatile int* cnt = bar + d * 64;
    int phase = 0;

    // initial barrier: all shards' h0 visible everywhere (also publishes LDS)
    __syncthreads();
    __threadfence();
    ++phase;
    if (tid == 0) {
        atomicAdd((int*)cnt, 1);
        while (*cnt < 32 * phase) __builtin_amdgcn_s_sleep(2);
    }
    __syncthreads();

    const int j = j0 + r;   // this lane's hidden unit

    for (int t = 0; t < TT; ++t) {
        const int tt = d ? (TT - 1 - t) : t;
        const __bf16* hrd = hbf + (size_t)(d * 2 + (t & 1)) * 16 * HHALF;
        __bf16*       hwr = hbf + (size_t)(d * 2 + ((t + 1) & 1)) * 16 * HHALF;

        // Issue the input-gate loads up front; they complete under the WMMAs.
        float xgv[4][8];
        const float* xrow = xg + (size_t)tt * BB * 4096;
#pragma unroll
        for (int g = 0; g < 4; ++g) {
            const int gcol = d * G4 + g * HHALF + j;
#pragma unroll
            for (int v = 0; v < 8; ++v)
                xgv[g][v] = xrow[(size_t)(half * 8 + v) * 4096 + gcol];
        }

        // gates = h @ w_hh^T  (A from global h double-buffer, B from LDS);
        // A-fragment loads are register-pipelined across k-steps.
        f32x8 acc[4];
#pragma unroll
        for (int g = 0; g < 4; ++g)
#pragma unroll
            for (int v = 0; v < 8; ++v) acc[g][v] = 0.0f;

        bf16x16 a_cur = load_frag(hrd, HHALF, r, 0, kb);
        for (int k0 = 0; k0 < HHALF; k0 += 32) {
            bf16x16 a_nxt;
            if (k0 + 32 < HHALF) a_nxt = load_frag(hrd, HHALF, r, k0 + 32, kb);
#pragma unroll
            for (int g = 0; g < 4; ++g) {
                FragAB bf;
                const __bf16* p = wl + (size_t)(g * 16 + r) * HHALF + k0 + kb;
                bf.q[0] = *(const int4*)p;
                bf.q[1] = *(const int4*)(p + 16);
                acc[g] = __builtin_amdgcn_wmma_f32_16x16x32_bf16(
                    false, a_cur, false, bf.v, (short)0, acc[g], false, false);
            }
            a_cur = a_nxt;
        }

        // i, f, g, o -> c, h   (PyTorch gate order)
#pragma unroll
        for (int v = 0; v < 8; ++v) {
            const int m = half * 8 + v;
            const float ig = sigmoidf_(acc[0][v] + xgv[0][v]);
            const float fg = sigmoidf_(acc[1][v] + xgv[1][v]);
            const float gg = tanhf(acc[2][v] + xgv[2][v]);
            const float og = sigmoidf_(acc[3][v] + xgv[3][v]);
            c[v] = fg * c[v] + ig * gg;
            const float h = og * tanhf(c[v]);
            enc[((size_t)m * TT + tt) * HID + d * HHALF + j] = h;
            hwr[(size_t)m * HHALF + j] = (__bf16)h;
        }

        // per-direction phase barrier
        __syncthreads();
        __threadfence();
        ++phase;
        if (tid == 0) {
            atomicAdd((int*)cnt, 1);
            while (*cnt < 32 * phase) __builtin_amdgcn_s_sleep(2);
        }
        __syncthreads();
        __threadfence();
    }
}

// ---------------------------------------------------------------------------
// Span mean pooling + LayerNorm. grid = B*S blocks, 256 threads.
// ---------------------------------------------------------------------------
__global__ __launch_bounds__(256)
void k_span_pool_ln(const float* __restrict__ enc, const int* __restrict__ heads,
                    const int* __restrict__ tails, const float* __restrict__ g,
                    const float* __restrict__ bta, float* __restrict__ feat,
                    __bf16* __restrict__ featb) {
    __shared__ float r1[256], r2[256];
    const int bs = blockIdx.x;
    const int b = bs >> 5;
    const int head = heads[bs], tail = tails[bs];
    const int lo = head + 1;
    const float inv = 1.0f / (float)max(tail - lo, 1);
    const int tid = threadIdx.x;

    float val[4];
#pragma unroll
    for (int k = 0; k < 4; ++k) {
        const int ch = tid + k * 256;
        float a = 0.0f;
        for (int tok = lo; tok < tail; ++tok)
            a += enc[((size_t)b * TT + tok) * HID + ch];
        val[k] = a * inv;
    }
    float s1 = 0.0f, s2 = 0.0f;
#pragma unroll
    for (int k = 0; k < 4; ++k) { s1 += val[k]; s2 += val[k] * val[k]; }
    r1[tid] = s1; r2[tid] = s2;
    __syncthreads();
    for (int o = 128; o > 0; o >>= 1) {
        if (tid < o) { r1[tid] += r1[tid + o]; r2[tid] += r2[tid + o]; }
        __syncthreads();
    }
    const float mu = r1[0] * (1.0f / HID);
    const float var = r2[0] * (1.0f / HID) - mu * mu;
    const float rs = rsqrtf(var + LN_EPS);
#pragma unroll
    for (int k = 0; k < 4; ++k) {
        const int ch = tid + k * 256;
        const float y = (val[k] - mu) * rs * g[ch] + bta[ch];
        feat[(size_t)bs * HID + ch] = y;
        featb[(size_t)bs * HID + ch] = (__bf16)y;
    }
}

// ---------------------------------------------------------------------------
// Attention scores: one thread per (b, h, q, k)
// ---------------------------------------------------------------------------
__global__ void k_scores(const float* __restrict__ qkv, const int* __restrict__ am,
                         float* __restrict__ scores) {
    const int idx = blockIdx.x * blockDim.x + threadIdx.x;
    if (idx >= BB * NHEAD * SS * SS) return;
    const int k = idx & 31, q = (idx >> 5) & 31, h = (idx >> 10) & 15, b = idx >> 14;
    const float* Q  = qkv + (size_t)(b * SS + q) * 3072 + h * DHEAD;
    const float* Kp = qkv + (size_t)(b * SS + k) * 3072 + 1024 + h * DHEAD;
    float sc = 0.0f;
#pragma unroll 8
    for (int d = 0; d < DHEAD; ++d) sc += Q[d] * Kp[d];
    sc *= 0.125f; // 1/sqrt(64)
    if (!(am[b * SS + q] && am[b * SS + k])) sc = -3.4028234663852886e38f;
    scores[idx] = sc;
}

// Softmax over k: one wave32 per (b,h,q) row. shuffle reductions.
__global__ __launch_bounds__(256)
void k_softmax(const float* __restrict__ scores, const int* __restrict__ am,
               float* __restrict__ probs) {
    const int row = blockIdx.x * 8 + (threadIdx.x >> 5);
    const int lane = threadIdx.x & 31;
    float v = scores[(size_t)row * SS + lane];
    float m = v;
#pragma unroll
    for (int o = 16; o > 0; o >>= 1) m = fmaxf(m, __shfl_xor(m, o, 32));
    float e = __expf(v - m);
    float sum = e;
#pragma unroll
    for (int o = 16; o > 0; o >>= 1) sum += __shfl_xor(sum, o, 32);
    float p = e / sum;
    const int b = row >> 9, q = row & 31;
    if (!(am[b * SS + q] && am[b * SS + lane])) p = 0.0f;
    probs[(size_t)row * SS + lane] = p;
}

// ctx = probs @ V : one thread per (b, s, ch)
__global__ void k_ctx(const float* __restrict__ probs, const float* __restrict__ qkv,
                      float* __restrict__ ctx, __bf16* __restrict__ ctxb) {
    const int idx = blockIdx.x * blockDim.x + threadIdx.x;
    if (idx >= BB * SS * HID) return;
    const int ch = idx & 1023, s = (idx >> 10) & 31, b = idx >> 15;
    const int h = ch >> 6;
    const float* pr = probs + ((size_t)(b * NHEAD + h) * SS + s) * SS;
    float a = 0.0f;
#pragma unroll 8
    for (int k = 0; k < SS; ++k)
        a += pr[k] * qkv[(size_t)(b * SS + k) * 3072 + 2048 + ch];
    ctx[idx] = a;
    ctxb[idx] = (__bf16)a;
}

// attn_out = LN(proj + feat)
__global__ __launch_bounds__(256)
void k_add_ln(const float* __restrict__ x, const float* __restrict__ res,
              const float* __restrict__ g, const float* __restrict__ bta,
              float* __restrict__ y) {
    __shared__ float r1[256], r2[256];
    const int row = blockIdx.x, tid = threadIdx.x;
    float val[4];
#pragma unroll
    for (int k = 0; k < 4; ++k) {
        const int ch = tid + k * 256;
        val[k] = x[(size_t)row * HID + ch] + res[(size_t)row * HID + ch];
    }
    float s1 = 0.0f, s2 = 0.0f;
#pragma unroll
    for (int k = 0; k < 4; ++k) { s1 += val[k]; s2 += val[k] * val[k]; }
    r1[tid] = s1; r2[tid] = s2;
    __syncthreads();
    for (int o = 128; o > 0; o >>= 1) {
        if (tid < o) { r1[tid] += r1[tid + o]; r2[tid] += r2[tid + o]; }
        __syncthreads();
    }
    const float mu = r1[0] * (1.0f / HID);
    const float var = r2[0] * (1.0f / HID) - mu * mu;
    const float rs = rsqrtf(var + LN_EPS);
#pragma unroll
    for (int k = 0; k < 4; ++k) {
        const int ch = tid + k * 256;
        y[(size_t)row * HID + ch] = (val[k] - mu) * rs * g[ch] + bta[ch];
    }
}

// logits = attn_out @ wc^T + bc : one thread per (row, label)
__global__ void k_classifier(const float* __restrict__ x, const float* __restrict__ wc,
                             const float* __restrict__ bc, float* __restrict__ out) {
    const int idx = blockIdx.x * blockDim.x + threadIdx.x;
    if (idx >= BB * SS * NL) return;
    const int row = idx / NL, l = idx - row * NL;
    const float* w = wc + (size_t)l * HID;
    const float* a = x + (size_t)row * HID;
    float s = 0.0f;
#pragma unroll 8
    for (int c = 0; c < HID; ++c) s += a[c] * w[c];
    out[idx] = s + bc[l];
}

// ---------------------------------------------------------------------------
// Host launcher
// ---------------------------------------------------------------------------
extern "C" void kernel_launch(void* const* d_in, const int* in_sizes, int n_in,
                              void* d_out, int out_size, void* d_ws, size_t ws_size,
                              hipStream_t stream) {
    (void)in_sizes; (void)n_in; (void)out_size; (void)ws_size;
    const float* x      = (const float*)d_in[0];
    const float* w_ih_f = (const float*)d_in[1];
    const float* w_hh_f = (const float*)d_in[2];
    const float* b_f    = (const float*)d_in[3];
    const float* w_ih_b = (const float*)d_in[4];
    const float* w_hh_b = (const float*)d_in[5];
    const float* b_b    = (const float*)d_in[6];
    const float* ln_g   = (const float*)d_in[7];
    const float* ln_b   = (const float*)d_in[8];
    const float* wq     = (const float*)d_in[9];
    const float* bq     = (const float*)d_in[10];
    const float* wk     = (const float*)d_in[11];
    const float* bk     = (const float*)d_in[12];
    const float* wv     = (const float*)d_in[13];
    const float* bv     = (const float*)d_in[14];
    const float* wo     = (const float*)d_in[15];
    const float* bo     = (const float*)d_in[16];
    const float* aln_g  = (const float*)d_in[17];
    const float* aln_b  = (const float*)d_in[18];
    const float* wc     = (const float*)d_in[19];
    const float* bc     = (const float*)d_in[20];
    const int* heads    = (const int*)d_in[21];
    const int* tails    = (const int*)d_in[22];
    const int* amask    = (const int*)d_in[23];
    float* out = (float*)d_out;

    // ---- workspace carve-up ----
    char* w = (char*)d_ws;
    size_t cur = 0;
    auto alloc = [&](size_t bytes) -> char* {
        char* p = w + cur;
        cur = (cur + bytes + 255) & ~(size_t)255;
        return p;
    };
    float*   xg        = (float*)  alloc((size_t)BB * TT * 4096 * 4);   // 256 MB, [T,B,4096]
    float*   enc       = (float*)  alloc((size_t)BB * TT * HID * 4);    //  64 MB
    __bf16*  x_bf      = (__bf16*) alloc((size_t)BB * TT * HID * 2);    //  32 MB
    __bf16*  wih_bf    = (__bf16*) alloc((size_t)4096 * HID * 2);       //   8 MB
    __bf16*  whh_bf    = (__bf16*) alloc((size_t)2 * G4 * HHALF * 2);   //   4 MB
    float*   bias_ih   = (float*)  alloc(4096 * 4);
    __bf16*  hbf       = (__bf16*) alloc((size_t)2 * 2 * 16 * HHALF * 2);
    int*     bar       = (int*)    alloc(512);
    float*   feat      = (float*)  alloc((size_t)BB * SS * HID * 4);
    __bf16*  feat_bf   = (__bf16*) alloc((size_t)BB * SS * HID * 2);
    __bf16*  wqkv_bf   = (__bf16*) alloc((size_t)3 * HID * HID * 2);
    float*   bias_qkv  = (float*)  alloc(3 * HID * 4);
    float*   qkv       = (float*)  alloc((size_t)BB * SS * 3 * HID * 4);
    float*   scores    = (float*)  alloc((size_t)BB * NHEAD * SS * SS * 4);
    float*   probs     = (float*)  alloc((size_t)BB * NHEAD * SS * SS * 4);
    float*   ctx       = (float*)  alloc((size_t)BB * SS * HID * 4);
    __bf16*  ctx_bf    = (__bf16*) alloc((size_t)BB * SS * HID * 2);
    __bf16*  wo_bf     = (__bf16*) alloc((size_t)HID * HID * 2);
    float*   proj      = (float*)  alloc((size_t)BB * SS * HID * 4);
    float*   attn_out  = (float*)  alloc((size_t)BB * SS * HID * 4);

    // ---- 0. reset barrier counters (deterministic across graph replays) ----
    hipMemsetAsync(bar, 0, 512, stream);

    // ---- 1. bf16 conversions of activations + weights ----
    const dim3 cb(256), cg(2048);
    k_f32_to_bf16<<<cg, cb, 0, stream>>>(x, x_bf, (long)BB * TT * HID);
    k_f32_to_bf16<<<cg, cb, 0, stream>>>(w_ih_f, wih_bf, (long)G4 * HID);
    k_f32_to_bf16<<<cg, cb, 0, stream>>>(w_ih_b, wih_bf + (size_t)G4 * HID, (long)G4 * HID);
    k_f32_to_bf16<<<cg, cb, 0, stream>>>(w_hh_f, whh_bf, (long)G4 * HHALF);
    k_f32_to_bf16<<<cg, cb, 0, stream>>>(w_hh_b, whh_bf + (size_t)G4 * HHALF, (long)G4 * HHALF);
    k_f32_to_bf16<<<cg, cb, 0, stream>>>(wq, wqkv_bf, (long)HID * HID);
    k_f32_to_bf16<<<cg, cb, 0, stream>>>(wk, wqkv_bf + (size_t)HID * HID, (long)HID * HID);
    k_f32_to_bf16<<<cg, cb, 0, stream>>>(wv, wqkv_bf + (size_t)2 * HID * HID, (long)HID * HID);
    k_f32_to_bf16<<<cg, cb, 0, stream>>>(wo, wo_bf, (long)HID * HID);
    hipMemcpyAsync(bias_ih,        b_f, G4 * 4, hipMemcpyDeviceToDevice, stream);
    hipMemcpyAsync(bias_ih + G4,   b_b, G4 * 4, hipMemcpyDeviceToDevice, stream);
    hipMemcpyAsync(bias_qkv,       bq, HID * 4, hipMemcpyDeviceToDevice, stream);
    hipMemcpyAsync(bias_qkv + HID, bk, HID * 4, hipMemcpyDeviceToDevice, stream);
    hipMemcpyAsync(bias_qkv + 2 * HID, bv, HID * 4, hipMemcpyDeviceToDevice, stream);

    // ---- 2. xg = x @ [w_ih_f; w_ih_b]^T + bias, stored [T, B, 4096] ----
    k_wmma_gemm<<<dim3(4096 / 256, (BB * TT) / 32), 128, 0, stream>>>(
        x_bf, wih_bf, bias_ih, xg, BB * TT, 4096, HID, 1);

    // ---- 3. bi-LSTM recurrence (persistent, LDS-resident w_hh shards) ----
    k_lstm_recurrence<<<dim3(32, 2), 128, 0, stream>>>(xg, whh_bf, hbf, enc, bar);

    // ---- 4. span mean-pool + LN ----
    k_span_pool_ln<<<BB * SS, 256, 0, stream>>>(enc, heads, tails, ln_g, ln_b,
                                                feat, feat_bf);

    // ---- 5. fused QKV projection (M=512, N=3072, K=1024) ----
    k_wmma_gemm<<<dim3(3072 / 256, (BB * SS) / 32), 128, 0, stream>>>(
        feat_bf, wqkv_bf, bias_qkv, qkv, BB * SS, 3 * HID, HID, 0);

    // ---- 6. attention core ----
    k_scores<<<(BB * NHEAD * SS * SS + 255) / 256, 256, 0, stream>>>(qkv, amask, scores);
    k_softmax<<<(BB * NHEAD * SS) / 8, 256, 0, stream>>>(scores, amask, probs);
    k_ctx<<<(BB * SS * HID + 255) / 256, 256, 0, stream>>>(probs, qkv, ctx, ctx_bf);

    // ---- 7. output projection + residual LN ----
    k_wmma_gemm<<<dim3(HID / 256, (BB * SS) / 32), 128, 0, stream>>>(
        ctx_bf, wo_bf, bo, proj, BB * SS, HID, HID, 0);
    k_add_ln<<<BB * SS, 256, 0, stream>>>(proj, feat, aln_g, aln_b, attn_out);

    // ---- 8. classifier ----
    k_classifier<<<(BB * SS * NL + 127) / 128, 128, 0, stream>>>(attn_out, wc, bc, out);
}
